// PtrNet_66073776882267
// MI455X (gfx1250) — matrix operand order repaired and checked by reading, not compiled, not run on hardware
//
#include <hip/hip_runtime.h>
#include <hip/hip_bf16.h>

// Problem constants
#define BB 256      // batch
#define SS 128      // sequence length
#define DD 2        // input dim
#define HH 512      // hidden
#define G4 2048     // 4*H (gates)

typedef __attribute__((ext_vector_type(16))) __bf16 v16bf;
typedef __attribute__((ext_vector_type(8)))  __bf16 v8bf;
typedef __attribute__((ext_vector_type(8)))  float  v8f;

__device__ __forceinline__ unsigned short f2bf(float f) {
    unsigned int u = __float_as_uint(f);
    if ((u & 0x7F800000u) == 0x7F800000u) return (unsigned short)(u >> 16); // inf/nan
    unsigned int r = u + 0x7FFFu + ((u >> 16) & 1u);                        // RNE
    return (unsigned short)(r >> 16);
}
__device__ __forceinline__ float sigmf(float x) { return 1.0f / (1.0f + __expf(-x)); }

// ---------------------------------------------------------------------------
// Pack fp32 weight W[NO][K] into bf16 WMMA B-matrix fragments.
// Fragment order: tile (jt over NO/16, kt over K/32), lane l (0..31), elem e (0..15)
//   B element = (K = kt*32 + 8*(l>>4) + (e&7) + 16*(e>>3), N = jt*16 + (l&15)) = W[N][K]
// ---------------------------------------------------------------------------
__global__ void pack_wfrag(const float* __restrict__ W, unsigned short* __restrict__ frag,
                           int NO, int K) {
    int i = blockIdx.x * blockDim.x + threadIdx.x;
    if (i >= NO * K) return;
    int e    = i & 15;
    int l    = (i >> 4) & 31;
    int tile = i >> 9;
    int kchunks = K >> 5;
    int kt = tile % kchunks;
    int jt = tile / kchunks;
    int n = jt * 16 + (l & 15);
    int k = kt * 32 + 8 * (l >> 4) + (e & 7) + 16 * (e >> 3);
    frag[i] = f2bf(W[(size_t)n * K + k]);
}

// zconst[j] = dec_bih[j] + dec_bhh[j] - sum_k dec_Wih[j][k]   (decoder input == -1)
__global__ void zconst_kernel(const float* __restrict__ Wih, const float* __restrict__ bih,
                              const float* __restrict__ bhh, float* __restrict__ zc) {
    int j = blockIdx.x * blockDim.x + threadIdx.x;
    if (j >= G4) return;
    float s = 0.f;
    const float* row = Wih + (size_t)j * HH;
    for (int k = 0; k < HH; ++k) s += row[k];
    zc[j] = bih[j] + bhh[j] - s;
}

__global__ void zero_state(float* __restrict__ c, unsigned short* __restrict__ hbf) {
    int i = blockIdx.x * blockDim.x + threadIdx.x;
    if (i >= BB * HH) return;
    c[i] = 0.f;
    hbf[i] = 0;
}

// ---------------------------------------------------------------------------
// C[256][N] = A_bf16[256][K] @ Wfrag (+ bias), fp32 accumulate via WMMA bf16.
// 2x2 register blocking, K fully unrolled (K is compile-time): each wave
// computes a 32x32 patch; per K-chunk 2 A + 2 B fragments feed 4 independent
// WMMAs. Full unroll + immediate offsets let the scheduler hoist loads across
// chunks and emit partial s_wait_loadcnt instead of full drains.
// ---------------------------------------------------------------------------
template <int K>
__global__ void gemm_bf16_256(const unsigned short* __restrict__ Abf,
                              const unsigned short* __restrict__ Wfrag,
                              const float* __restrict__ bias,
                              float* __restrict__ out, int N) {
    const int gtid = blockIdx.x * blockDim.x + threadIdx.x;
    const int wave = gtid >> 5;
    const int lane = threadIdx.x & 31;
    const int nblk = N >> 5;                  // 32-column blocks
    const int mb = wave / nblk;               // 32-row block index (0..7)
    const int nb = wave - mb * nblk;
    if (mb >= (BB >> 5)) return;

    const int half = lane >> 4;
    constexpr int kchunks = K >> 5;

    const __bf16* A = (const __bf16*)Abf;
    const __bf16* W = (const __bf16*)Wfrag;

    v8f acc00 = {}, acc01 = {}, acc10 = {}, acc11 = {};

    const __bf16* arow0 = A + (size_t)(mb * 32 + (lane & 15)) * K + 8 * half;
    const __bf16* arow1 = arow0 + (size_t)16 * K;
    const __bf16* wf0   = W + ((size_t)((nb * 2 + 0) * kchunks) * 32 + lane) * 16;
    const __bf16* wf1   = W + ((size_t)((nb * 2 + 1) * kchunks) * 32 + lane) * 16;

#pragma unroll
    for (int kt = 0; kt < kchunks; ++kt) {
        union { v16bf v; v8bf h[2]; } a0, a1;
        a0.h[0] = *(const v8bf*)(arow0 + kt * 32);
        a0.h[1] = *(const v8bf*)(arow0 + kt * 32 + 16);
        a1.h[0] = *(const v8bf*)(arow1 + kt * 32);
        a1.h[1] = *(const v8bf*)(arow1 + kt * 32 + 16);
        v16bf b0 = *(const v16bf*)(wf0 + kt * 512);
        v16bf b1 = *(const v16bf*)(wf1 + kt * 512);
        acc00 = __builtin_amdgcn_wmma_f32_16x16x32_bf16(false, a0.v, false, b0,
                                                        (short)0, acc00, false, false);
        acc01 = __builtin_amdgcn_wmma_f32_16x16x32_bf16(false, a0.v, false, b1,
                                                        (short)0, acc01, false, false);
        acc10 = __builtin_amdgcn_wmma_f32_16x16x32_bf16(false, a1.v, false, b0,
                                                        (short)0, acc10, false, false);
        acc11 = __builtin_amdgcn_wmma_f32_16x16x32_bf16(false, a1.v, false, b1,
                                                        (short)0, acc11, false, false);
    }

    const int col  = (nb << 5) + (lane & 15);      // first of two 16-col tiles
    const int row0 = (mb << 5) + 8 * half;         // first of two 16-row tiles
    const float bs0 = bias ? bias[col]      : 0.0f;
    const float bs1 = bias ? bias[col + 16] : 0.0f;
#pragma unroll
    for (int r = 0; r < 8; ++r) {
        out[(size_t)(row0 + r)      * N + col]      = acc00[r] + bs0;
        out[(size_t)(row0 + r)      * N + col + 16] = acc01[r] + bs1;
        out[(size_t)(row0 + 16 + r) * N + col]      = acc10[r] + bs0;
        out[(size_t)(row0 + 16 + r) * N + col + 16] = acc11[r] + bs1;
    }
}

// ---------------------------------------------------------------------------
// Encoder LSTM cell: gates from z (=h@Whh.T) + x_t@Wih.T + biases.
// Writes c, h (fp32, into enc_hs[t]) and h as bf16 for the next GEMM.
// ---------------------------------------------------------------------------
__global__ void enc_gate(const float* __restrict__ z, const float* __restrict__ x,
                         const float* __restrict__ Wih, const float* __restrict__ bih,
                         const float* __restrict__ bhh, float* __restrict__ c,
                         float* __restrict__ h_out, unsigned short* __restrict__ hbf, int t) {
    int i = blockIdx.x * blockDim.x + threadIdx.x;
    if (i >= BB * HH) return;
    int b = i / HH, n = i - b * HH;
    float x0 = x[((size_t)b * SS + t) * DD + 0];
    float x1 = x[((size_t)b * SS + t) * DD + 1];
    float zg[4];
#pragma unroll
    for (int g = 0; g < 4; ++g) {
        int j = g * HH + n;
        zg[g] = z[(size_t)b * G4 + j] + x0 * Wih[j * DD + 0] + x1 * Wih[j * DD + 1]
              + bih[j] + bhh[j];
    }
    float cn = sigmf(zg[1]) * c[i] + sigmf(zg[0]) * tanhf(zg[2]);
    float hn = sigmf(zg[3]) * tanhf(cn);
    c[i] = cn;
    h_out[i] = hn;
    hbf[i] = f2bf(hn);
}

// Decoder LSTM cell: z + precomputed zconst (input term + biases).
__global__ void dec_gate(const float* __restrict__ z, const float* __restrict__ zc,
                         float* __restrict__ c, unsigned short* __restrict__ hbf) {
    int i = blockIdx.x * blockDim.x + threadIdx.x;
    if (i >= BB * HH) return;
    int b = i / HH, n = i - b * HH;
    float zg[4];
#pragma unroll
    for (int g = 0; g < 4; ++g) {
        int j = g * HH + n;
        zg[g] = z[(size_t)b * G4 + j] + zc[j];
    }
    float cn = sigmf(zg[1]) * c[i] + sigmf(zg[0]) * tanhf(zg[2]);
    float hn = sigmf(zg[3]) * tanhf(cn);
    c[i] = cn;
    hbf[i] = f2bf(hn);
}

// u[b][s] = v . tanh(W1e[s][b][:] + W2h[b][:]) + bv    (one wave per (b,s))
__global__ void score_kernel(const float* __restrict__ W1e, const float* __restrict__ W2h,
                             const float* __restrict__ v, const float* __restrict__ bv,
                             float* __restrict__ u) {
    int gtid = blockIdx.x * blockDim.x + threadIdx.x;
    int wave = gtid >> 5, lane = threadIdx.x & 31;
    int b = wave / SS, s = wave - b * SS;
    if (b >= BB) return;
    const float* we = W1e + ((size_t)s * BB + b) * HH;
    const float* wh = W2h + (size_t)b * HH;
    float acc = 0.f;
#pragma unroll 4
    for (int e = 0; e < HH / 32; ++e) {
        int idx = e * 32 + lane;
        acc += tanhf(we[idx] + wh[idx]) * v[idx];
    }
    for (int m = 16; m >= 1; m >>= 1) acc += __shfl_xor(acc, m, 32);
    if (lane == 0) u[wave] = acc + bv[0];
}

// Softmax over S, write attention row to d_out, compute glimpse -> next h (bf16).
__global__ void softmax_glimpse(const float* __restrict__ u, const float* __restrict__ enc_hs,
                                float* __restrict__ out, unsigned short* __restrict__ hbf, int t) {
    const int b = blockIdx.x;
    const int s = threadIdx.x;            // 128 threads
    __shared__ float sh[SS];
    __shared__ float red[SS];
    float val = u[b * SS + s];
    red[s] = val; __syncthreads();
    for (int off = 64; off >= 1; off >>= 1) {
        if (s < off) red[s] = fmaxf(red[s], red[s + off]);
        __syncthreads();
    }
    float mx = red[0]; __syncthreads();
    float e = __expf(val - mx);
    red[s] = e; __syncthreads();
    for (int off = 64; off >= 1; off >>= 1) {
        if (s < off) red[s] += red[s + off];
        __syncthreads();
    }
    float a = e / red[0];
    sh[s] = a;
    out[(size_t)b * SS * SS + (size_t)t * SS + s] = a;
    __syncthreads();
    for (int h = s; h < HH; h += SS) {
        float acc = 0.f;
        for (int sp = 0; sp < SS; ++sp)
            acc += sh[sp] * enc_hs[((size_t)sp * BB + b) * HH + h];
        hbf[b * HH + h] = f2bf(acc);
    }
}

// ---------------------------------------------------------------------------
extern "C" void kernel_launch(void* const* d_in, const int* in_sizes, int n_in,
                              void* d_out, int out_size, void* d_ws, size_t ws_size,
                              hipStream_t stream) {
    const float* x        = (const float*)d_in[0];
    const float* enc_Wih  = (const float*)d_in[1];
    const float* enc_Whh  = (const float*)d_in[2];
    const float* enc_bih  = (const float*)d_in[3];
    const float* enc_bhh  = (const float*)d_in[4];
    const float* dec_Wih  = (const float*)d_in[5];
    const float* dec_Whh  = (const float*)d_in[6];
    const float* dec_bih  = (const float*)d_in[7];
    const float* dec_bhh  = (const float*)d_in[8];
    const float* W1       = (const float*)d_in[9];
    const float* b1       = (const float*)d_in[10];
    const float* W2       = (const float*)d_in[11];
    const float* b2       = (const float*)d_in[12];
    const float* v        = (const float*)d_in[13];
    const float* bv       = (const float*)d_in[14];
    float* out = (float*)d_out;

    // ---- workspace carving (256B aligned) ----
    uintptr_t p = (uintptr_t)d_ws;
    auto take = [&](size_t bytes) -> void* {
        void* r = (void*)p;
        p += (bytes + 255) & ~(size_t)255;
        return r;
    };
    float*          enc_hs   = (float*)take((size_t)SS * BB * HH * 4);   // [t][B][H]
    float*          W1e      = (float*)take((size_t)SS * BB * HH * 4);   // [t][B][H]
    float*          zbuf     = (float*)take((size_t)BB * G4 * 4);        // gate preacts
    float*          cbuf     = (float*)take((size_t)BB * HH * 4);        // cell state
    unsigned short* hbf      = (unsigned short*)take((size_t)BB * HH * 2);   // recurrent h (bf16)
    unsigned short* hlstm_bf = (unsigned short*)take((size_t)BB * HH * 2);   // decoder LSTM h (bf16)
    float*          W2h      = (float*)take((size_t)BB * HH * 4);
    float*          ubuf     = (float*)take((size_t)BB * SS * 4);
    unsigned short* encWhh_f = (unsigned short*)take((size_t)G4 * HH * 2);
    unsigned short* decWhh_f = (unsigned short*)take((size_t)G4 * HH * 2);
    unsigned short* W1_f     = (unsigned short*)take((size_t)HH * HH * 2);
    unsigned short* W2_f     = (unsigned short*)take((size_t)HH * HH * 2);
    float*          zconst   = (float*)take((size_t)G4 * 4);

    // ---- one-time prep (runs every call; deterministic) ----
    {
        int nBig = G4 * HH;                       // 1M elements
        pack_wfrag<<<(nBig + 255) / 256, 256, 0, stream>>>(enc_Whh, encWhh_f, G4, HH);
        pack_wfrag<<<(nBig + 255) / 256, 256, 0, stream>>>(dec_Whh, decWhh_f, G4, HH);
        int nSml = HH * HH;                       // 256K elements
        pack_wfrag<<<(nSml + 255) / 256, 256, 0, stream>>>(W1, W1_f, HH, HH);
        pack_wfrag<<<(nSml + 255) / 256, 256, 0, stream>>>(W2, W2_f, HH, HH);
        zconst_kernel<<<(G4 + 255) / 256, 256, 0, stream>>>(dec_Wih, dec_bih, dec_bhh, zconst);
        zero_state<<<(BB * HH + 255) / 256, 256, 0, stream>>>(cbuf, hbf);
    }

    // 2x2-blocked GEMM: waves = (256/32) * (N/32); 8 waves per 256-thread block
    const int blkZ  = ((BB / 32) * (G4 / 32)) / 8;   // 512 waves -> 64 blocks
    const int blkH  = ((BB / 32) * (HH / 32)) / 8;   // 128 waves -> 16 blocks
    const int blkBH = (BB * HH + 255) / 256;         // 512
    const int blkSc = (BB * SS * 32) / 256;          // 4096

    // ---- encoder ----
    for (int t = 0; t < SS; ++t) {
        gemm_bf16_256<HH><<<blkZ, 256, 0, stream>>>(hbf, encWhh_f, nullptr, zbuf, G4);
        enc_gate<<<blkBH, 256, 0, stream>>>(zbuf, x, enc_Wih, enc_bih, enc_bhh,
                                            cbuf, enc_hs + (size_t)t * BB * HH, hbf, t);
        gemm_bf16_256<HH><<<blkH, 256, 0, stream>>>(hbf, W1_f, b1,
                                                    W1e + (size_t)t * BB * HH, HH);
    }

    // ---- decoder with attention (h carried in hbf, c in cbuf) ----
    for (int t = 0; t < SS; ++t) {
        gemm_bf16_256<HH><<<blkZ, 256, 0, stream>>>(hbf, decWhh_f, nullptr, zbuf, G4);
        dec_gate<<<blkBH, 256, 0, stream>>>(zbuf, zconst, cbuf, hlstm_bf);
        gemm_bf16_256<HH><<<blkH, 256, 0, stream>>>(hlstm_bf, W2_f, b2, W2h, HH);
        score_kernel<<<blkSc, 256, 0, stream>>>(W1e, W2h, v, bv, ubuf);
        softmax_glimpse<<<BB, SS, 0, stream>>>(ubuf, enc_hs, out, hbf, t);
    }
}
